// SelectiveSSMCore_12610023981604
// MI455X (gfx1250) — matrix operand order repaired and checked by reading, not compiled, not run on hardware
//
#include <hip/hip_runtime.h>
#include <math.h>

// ---------------------------------------------------------------------------
// Problem constants (match reference)
// ---------------------------------------------------------------------------
#define BB 2
#define LL 2048
#define DD 1024
#define NN 16
#define MM (BB * LL)          // 4096 GEMM rows
#define KK DD                 // 1024 GEMM reduction dim
#define NOUT_BC (DD * NN)     // 16384
#define NOUT_DT DD            // 1024
#define TCH 32                // scan chunk (timesteps staged per TDM load)

// ---------------------------------------------------------------------------
// Vector types
// ---------------------------------------------------------------------------
typedef __bf16    v4bf  __attribute__((ext_vector_type(4)));
typedef __bf16    v8bf  __attribute__((ext_vector_type(8)));
typedef __bf16    v16bf __attribute__((ext_vector_type(16)));
typedef float     v8f   __attribute__((ext_vector_type(8)));
typedef unsigned  v4u   __attribute__((ext_vector_type(4)));
typedef unsigned  v8u   __attribute__((ext_vector_type(8)));

// ---------------------------------------------------------------------------
// fp32 -> bf16 conversion (4 elements / thread)
// ---------------------------------------------------------------------------
__global__ __launch_bounds__(256) void cvt_f32_to_bf16_kernel(
    const float* __restrict__ src, __bf16* __restrict__ dst, int n4)
{
  int i = blockIdx.x * blockDim.x + threadIdx.x;
  if (i < n4) {
    float4 v = reinterpret_cast<const float4*>(src)[i];
    v4bf o;
    o[0] = (__bf16)v.x; o[1] = (__bf16)v.y; o[2] = (__bf16)v.z; o[3] = (__bf16)v.w;
    reinterpret_cast<v4bf*>(dst)[i] = o;
  }
}

// ---------------------------------------------------------------------------
// WMMA fragment loaders (CDNA5 ISA 7.12.2, 16-bit operands, wave32)
// ---------------------------------------------------------------------------
__device__ __forceinline__ v16bf load_frag_a(const __bf16* __restrict__ rowptr,
                                             int k, int lh)
{
  const __bf16* p0 = rowptr + k + lh * 8;
  v8bf lo = *reinterpret_cast<const v8bf*>(p0);
  v8bf hi = *reinterpret_cast<const v8bf*>(p0 + 16);
  return __builtin_shufflevector(lo, hi, 0,1,2,3,4,5,6,7,8,9,10,11,12,13,14,15);
}

__device__ __forceinline__ v16bf load_frag_b(const __bf16* __restrict__ rowptr,
                                             int k, int lh)
{
  const __bf16* p0 = rowptr + k + lh * 8;
  v8bf lo = *reinterpret_cast<const v8bf*>(p0);
  v8bf hi = *reinterpret_cast<const v8bf*>(p0 + 16);
  return __builtin_shufflevector(lo, hi, 0,1,2,3,4,5,6,7,8,9,10,11,12,13,14,15);
}

// ---------------------------------------------------------------------------
// WMMA bf16 GEMM:  out[M, Nout] = X[M,K] @ W[Nout,K]^T + bias
//   MODE 0: fp32 out, softplus(acc + bias + bias2)   (dt path)
//   MODE 1: bf16 out, acc + bias                     (Bt / Ct path)
// Block = 256 threads = 8 waves (2M x 4N); wave tile 32x64; block tile 64x256.
// ---------------------------------------------------------------------------
template <int MODE>
__global__ __launch_bounds__(256) void wmma_gemm_kernel(
    const __bf16* __restrict__ X, const __bf16* __restrict__ W,
    const float* __restrict__ bias, const float* __restrict__ bias2,
    void* __restrict__ outv, int M, int Nout, int K)
{
  const int lane = threadIdx.x & 31;
  const int wid  = threadIdx.x >> 5;
  const int wm   = wid & 1;
  const int wn   = wid >> 1;
  const int lm   = lane & 15;
  const int lh   = lane >> 4;

  const int tileM = blockIdx.y * 64  + wm * 32;
  const int tileN = blockIdx.x * 256 + wn * 64;

  v8f acc[2][4];
#pragma unroll
  for (int mi = 0; mi < 2; ++mi)
#pragma unroll
    for (int ni = 0; ni < 4; ++ni)
      acc[mi][ni] = {};

  const __bf16* arow = X + (size_t)(tileM + lm) * K;
  const __bf16* brow = W + (size_t)(tileN + lm) * K;

  for (int k = 0; k < K; k += 32) {
    v16bf afrag[2], bfrag[4];
#pragma unroll
    for (int mi = 0; mi < 2; ++mi)
      afrag[mi] = load_frag_a(arow + (size_t)mi * 16 * K, k, lh);
#pragma unroll
    for (int ni = 0; ni < 4; ++ni)
      bfrag[ni] = load_frag_b(brow + (size_t)ni * 16 * K, k, lh);

#pragma unroll
    for (int mi = 0; mi < 2; ++mi)
#pragma unroll
      for (int ni = 0; ni < 4; ++ni)
        acc[mi][ni] = __builtin_amdgcn_wmma_f32_16x16x32_bf16(
            false, afrag[mi], false, bfrag[ni],
            (short)0, acc[mi][ni], false, false);
  }

#pragma unroll
  for (int mi = 0; mi < 2; ++mi) {
    const int rbase = tileM + mi * 16 + lh * 8;
#pragma unroll
    for (int ni = 0; ni < 4; ++ni) {
      const int col = tileN + ni * 16 + lm;
      float badd = bias[col];
      if (MODE == 0) badd += bias2[col];
#pragma unroll
      for (int v = 0; v < 8; ++v) {
        const float val = acc[mi][ni][v] + badd;
        const size_t oidx = (size_t)(rbase + v) * (size_t)Nout + (size_t)col;
        if (MODE == 0) {
          // fast softplus: exact enough for dt (|x| small); avoids the
          // precise log1p VALU expansion
          const float sp = (val > 15.0f) ? val
                                         : __logf(1.0f + __expf(val));
          reinterpret_cast<float*>(outv)[oidx] = sp;
        } else {
          reinterpret_cast<__bf16*>(outv)[oidx] = (__bf16)val;
        }
      }
    }
  }
}

// ---------------------------------------------------------------------------
// Tensor Data Mover: 2D tile load, descriptor per CDNA5 ISA ch.8 (D# groups).
//   lds_off       : byte offset of destination inside workgroup LDS
//   gptr          : global address of *tile start*
//   tensor_d0/d1  : tensor extents (elements / rows) for OOB clamping
//   tile_d0/d1    : tile extents (elements / rows)
//   stride0       : elements between consecutive rows
//   dsz_code      : 0=1B 1=2B 2=4B 3=8B
// All arguments are workgroup-uniform -> SGPRs for the "s" constraints.
// ---------------------------------------------------------------------------
__device__ __forceinline__ void tdm_load_2d(
    unsigned lds_off, const void* gptr,
    unsigned tensor_d0, unsigned tensor_d1,
    unsigned tile_d0, unsigned tile_d1,
    unsigned long long stride0, unsigned dsz_code)
{
  const unsigned long long ga = (unsigned long long)(uintptr_t)gptr;
  v4u g0;
  g0[0] = 1u;                                         // count=1, user mode
  g0[1] = lds_off;                                    // lds_addr (bytes)
  g0[2] = (unsigned)(ga & 0xFFFFFFFFull);             // global_addr[31:0]
  g0[3] = (unsigned)((ga >> 32) & 0x1FFFFFFull)       // global_addr[56:32]
        | (2u << 30);                                 // type = 2 ("image")
  v8u g1;
  g1[0] = (dsz_code & 3u) << 16;                      // data_size; wg_mask=0
  g1[1] = (tensor_d0 & 0xFFFFu) << 16;                // tensor_dim0[15:0]
  g1[2] = ((tensor_d0 >> 16) & 0xFFFFu)               // tensor_dim0[31:16]
        | ((tensor_d1 & 0xFFFFu) << 16);              // tensor_dim1[15:0]
  g1[3] = ((tensor_d1 >> 16) & 0xFFFFu)               // tensor_dim1[31:16]
        | ((tile_d0 & 0xFFFFu) << 16);                // tile_dim0
  g1[4] = (tile_d1 & 0xFFFFu);                        // tile_dim1; tile_dim2=0
  g1[5] = (unsigned)(stride0 & 0xFFFFFFFFull);        // dim0_stride[31:0]
  g1[6] = (unsigned)((stride0 >> 32) & 0xFFFFull);    // dim0_stride[47:32]
  g1[7] = 0u;                                         // dim1_stride unused (2D)
  asm volatile("tensor_load_to_lds %0, %1" :: "s"(g0), "s"(g1) : "memory");
}

__device__ __forceinline__ unsigned lds_offset_of(const void* p)
{
  // flat shared pointer: low 32 bits are the wave-relative LDS byte address
  return (unsigned)(uintptr_t)p;
}

// ---------------------------------------------------------------------------
// Sequential scan over L with TDM-staged operands.
// Block = 256 threads = 16 contiguous d-channels x 16 states (one b).
// Per chunk of TCH timesteps, wave 0 DMAs:
//   Bt tile : TCH x 256 bf16 (cols d0*N .. d0*N+256, stride D*N)
//   Ct tile : TCH x 256 bf16
//   dt tile : TCH x 16  f32  (cols d0 .. d0+16,     stride D)
//   u  tile : TCH x 16  f32
// into double-buffered LDS (72 KB total), waits TENSORcnt, barrier, compute.
// ---------------------------------------------------------------------------
__global__ __launch_bounds__(256) void ssm_scan_kernel(
    const float* __restrict__ dt, const __bf16* __restrict__ Bt,
    const __bf16* __restrict__ Ct, const float* __restrict__ u,
    const float* __restrict__ log_A, const float* __restrict__ D_skip,
    float* __restrict__ y)
{
  __shared__ __align__(16) __bf16 sB[2][TCH * 256];
  __shared__ __align__(16) __bf16 sC[2][TCH * 256];
  __shared__ __align__(16) float  sDt[2][TCH * 16];
  __shared__ __align__(16) float  sU [2][TCH * 16];

  const int tid  = threadIdx.x;
  const int n    = tid & 15;
  const int dloc = tid >> 4;                 // 0..15
  const int wid  = tid >> 5;
  const int bI   = blockIdx.x >> 6;          // batch (blocks 0..63 -> b=0)
  const int d0   = (blockIdx.x & 63) << 4;   // first d channel of this block
  const int d    = d0 + dloc;

  const float A   = -__expf(log_A[d * NN + n]);
  const float rA  = 1.0f / A;
  const float dsk = D_skip[d];

  const int nchunks = LL / TCH;

  // issue chunk 0 into buffer 0
  if (wid == 0) {
    const size_t row0 = (size_t)bI * LL;     // first tile row (l0 = 0)
    tdm_load_2d(lds_offset_of(&sB[0][0]),  Bt + row0 * NOUT_BC + (size_t)d0 * NN,
                NOUT_BC, MM, 256, TCH, NOUT_BC, 1u);
    tdm_load_2d(lds_offset_of(&sC[0][0]),  Ct + row0 * NOUT_BC + (size_t)d0 * NN,
                NOUT_BC, MM, 256, TCH, NOUT_BC, 1u);
    tdm_load_2d(lds_offset_of(&sDt[0][0]), dt + row0 * DD + d0,
                DD, MM, 16, TCH, DD, 2u);
    tdm_load_2d(lds_offset_of(&sU[0][0]),  u  + row0 * DD + d0,
                DD, MM, 16, TCH, DD, 2u);
  }

  float h = 0.0f;

  for (int c = 0; c < nchunks; ++c) {
    const int s = c & 1;
    if (wid == 0) {
      __builtin_amdgcn_s_wait_tensorcnt(0);  // buffer s ready
    }
    __syncthreads();                         // publish buffer s to all waves

    // prefetch next chunk into the other buffer (it was fully consumed in
    // chunk c-1; the barrier above makes that globally visible)
    if (wid == 0 && (c + 1) < nchunks) {
      const int sn = s ^ 1;
      const size_t rown = (size_t)(bI * LL + (c + 1) * TCH);
      tdm_load_2d(lds_offset_of(&sB[sn][0]),  Bt + rown * NOUT_BC + (size_t)d0 * NN,
                  NOUT_BC, MM, 256, TCH, NOUT_BC, 1u);
      tdm_load_2d(lds_offset_of(&sC[sn][0]),  Ct + rown * NOUT_BC + (size_t)d0 * NN,
                  NOUT_BC, MM, 256, TCH, NOUT_BC, 1u);
      tdm_load_2d(lds_offset_of(&sDt[sn][0]), dt + rown * DD + d0,
                  DD, MM, 16, TCH, DD, 2u);
      tdm_load_2d(lds_offset_of(&sU[sn][0]),  u  + rown * DD + d0,
                  DD, MM, 16, TCH, DD, 2u);
    }

    const int l0 = c * TCH;
#pragma unroll 4
    for (int t = 0; t < TCH; ++t) {
      const float dtv = sDt[s][t * 16 + dloc];   // broadcast within n-group
      const float uv  = sU [s][t * 16 + dloc];
      const float a   = __expf(dtv * A);
      const float bt  = (float)sB[s][t * 256 + tid];
      const float ct  = (float)sC[s][t * 256 + tid];
      const float x   = (a - 1.0f) * rA * bt * uv;
      h = fmaf(a, h, x);

      float p = ct * h;
      p += __shfl_xor(p, 8, 32);
      p += __shfl_xor(p, 4, 32);
      p += __shfl_xor(p, 2, 32);
      p += __shfl_xor(p, 1, 32);
      if (n == 0)
        y[(size_t)(bI * LL + l0 + t) * DD + d] = p + dsk * uv;
    }
    // no end-of-chunk barrier needed: the next iteration's top barrier
    // orders all reads of buffer s before any overwrite (issued at c+1)
  }
}

// ---------------------------------------------------------------------------
// Host-side launch. Inputs (setup_inputs order):
//  0 u (B,L,D) f32        1 log_A (D,N) f32     2 D_skip (D,) f32
//  3 dt_bias (D,) f32     4 W_dt (D,D) f32      5 b_dt (D,) f32
//  6 W_B (D*N,D) f32      7 b_B (D*N,) f32      8 W_C (D*N,D) f32
//  9 b_C (D*N,) f32
// Output: y (B,L,D) f32.
// Workspace layout: u_bf16 | Wdt_bf16 | WB_bf16 | WC_bf16 | dt f32 | Bt bf16 | Ct bf16
// (~346 MiB required)
// ---------------------------------------------------------------------------
extern "C" void kernel_launch(void* const* d_in, const int* in_sizes, int n_in,
                              void* d_out, int out_size, void* d_ws, size_t ws_size,
                              hipStream_t stream)
{
  const float* u       = (const float*)d_in[0];
  const float* log_A   = (const float*)d_in[1];
  const float* D_skip  = (const float*)d_in[2];
  const float* dt_bias = (const float*)d_in[3];
  const float* W_dt    = (const float*)d_in[4];
  const float* b_dt    = (const float*)d_in[5];
  const float* W_B     = (const float*)d_in[6];
  const float* b_B     = (const float*)d_in[7];
  const float* W_C     = (const float*)d_in[8];
  const float* b_C     = (const float*)d_in[9];
  float* y = (float*)d_out;

  char* ws = (char*)d_ws;
  const size_t n_u   = (size_t)MM * KK;
  const size_t n_wdt = (size_t)NOUT_DT * KK;
  const size_t n_wbc = (size_t)NOUT_BC * KK;
  const size_t n_btc = (size_t)MM * NOUT_BC;

  __bf16* u_bf   = (__bf16*)(ws);
  __bf16* wdt_bf = (__bf16*)(ws + n_u * 2);
  __bf16* wb_bf  = (__bf16*)(ws + n_u * 2 + n_wdt * 2);
  __bf16* wc_bf  = (__bf16*)(ws + n_u * 2 + n_wdt * 2 + n_wbc * 2);
  float*  dt_f   = (float*) (ws + n_u * 2 + n_wdt * 2 + n_wbc * 4);
  __bf16* bt_bf  = (__bf16*)(ws + n_u * 2 + n_wdt * 2 + n_wbc * 4 + n_u * 4);
  __bf16* ct_bf  = (__bf16*)(ws + n_u * 2 + n_wdt * 2 + n_wbc * 4 + n_u * 4 + n_btc * 2);

  // 1) fp32 -> bf16 conversions
  {
    const int blk = 256;
    int n4;
    n4 = (int)(n_u / 4);
    cvt_f32_to_bf16_kernel<<<(n4 + blk - 1) / blk, blk, 0, stream>>>(u, u_bf, n4);
    n4 = (int)(n_wdt / 4);
    cvt_f32_to_bf16_kernel<<<(n4 + blk - 1) / blk, blk, 0, stream>>>(W_dt, wdt_bf, n4);
    n4 = (int)(n_wbc / 4);
    cvt_f32_to_bf16_kernel<<<(n4 + blk - 1) / blk, blk, 0, stream>>>(W_B, wb_bf, n4);
    cvt_f32_to_bf16_kernel<<<(n4 + blk - 1) / blk, blk, 0, stream>>>(W_C, wc_bf, n4);
  }

  // 2) WMMA GEMMs
  {
    dim3 blk(256, 1, 1);
    dim3 g_dt(NOUT_DT / 256, MM / 64, 1);
    wmma_gemm_kernel<0><<<g_dt, blk, 0, stream>>>(
        u_bf, wdt_bf, b_dt, dt_bias, (void*)dt_f, MM, NOUT_DT, KK);
    dim3 g_bc(NOUT_BC / 256, MM / 64, 1);
    wmma_gemm_kernel<1><<<g_bc, blk, 0, stream>>>(
        u_bf, wb_bf, b_B, nullptr, (void*)bt_bf, MM, NOUT_BC, KK);
    wmma_gemm_kernel<1><<<g_bc, blk, 0, stream>>>(
        u_bf, wc_bf, b_C, nullptr, (void*)ct_bf, MM, NOUT_BC, KK);
  }

  // 3) recurrent scan + output (TDM-staged)
  {
    const int nblocks = BB * DD / 16;   // 128 blocks, 16 channels each
    ssm_scan_kernel<<<nblocks, 256, 0, stream>>>(
        dt_f, bt_bf, ct_bf, u, log_A, D_skip, y);
  }
}
